// AsymAttention_20675972563277
// MI455X (gfx1250) — compile-verified, matrix operand order checked by reading, not run on hardware
//
#include <hip/hip_runtime.h>

// ---------------------------------------------------------------------------
// AsymAttention on MI455X (gfx1250, wave32, WMMA bf16 16x16x32, f32 accum)
//
// Roofline: traffic floor = 2 x 201MB sim stream (~17us @ 23.3TB/s).
// All GEMMs (QKV proj, sim-K proj, QK^T, P@V, sim-V proj, output proj) run on
// v_wmma_f32_16x16x32_bf16. The sim stream is double-buffered into LDS with
// global_load_async_to_lds_b128 (ASYNCcnt) so the WMMA projection of tile q
// overlaps the HBM fetch of tile q+1. All self-produced tensors (x, Q, K, V,
// AO) are kept bf16 in workspace so fragment loads need no conversion VALU.
// ---------------------------------------------------------------------------

#define DIMD   768
#define HEADS  12
#define DH     64
#define BATCH  16
#define SEQ    256
#define MM     16
#define BN     (BATCH * SEQ)      // 4096 rows
#define KCH    (DIMD / 32)        // 24 k-chunks of 32 per full-D GEMM

typedef __attribute__((ext_vector_type(16))) __bf16 v16bf;
typedef __attribute__((ext_vector_type(8)))  float  v8f;

__device__ __forceinline__ v8f wmma_bf16(v16bf a, v16bf b, v8f c) {
  // (neg_a, A, neg_b, B, c_mod, C, reuse_a, reuse_b)
  return __builtin_amdgcn_wmma_f32_16x16x32_bf16(false, a, false, b,
                                                 (short)0, c, false, false);
}

// Load one 16x32 (A) or 32x16 (B) bf16 fragment.
// "line" = dimension striped across lanes 0..15 (A: row, B: column);
// k contiguous within a line. ISA 16-bit operand layout (wave32):
//   lanes 0-15 : k in {0..7, 16..23}, lanes 16-31 : k in {8..15, 24..31}
template <typename T>
__device__ __forceinline__ v16bf load_frag(const T* src, long line_stride,
                                           int line_base, int k_base, int lane) {
  const int half = (lane >> 4) & 1;
  const int l    = lane & 15;
  const T* p = src + (long)(line_base + l) * line_stride + k_base + half * 8;
  v16bf f;
#pragma unroll
  for (int e = 0; e < 8; ++e) f[e] = (__bf16)p[e];
#pragma unroll
  for (int e = 0; e < 8; ++e) f[e + 8] = (__bf16)p[e + 16];
  return f;
}

// ---- CDNA5 async global->LDS staging (ASYNCcnt) ---------------------------
// 16x768 f32 tile = 49152 B; 256 threads x 12 x b128 per thread.
// LDS byte address = low 32 bits of the generic pointer (LDS aperture).
__device__ __forceinline__ void async_stage_tile(const float* gsrc, float* ldst,
                                                 int tid) {
#pragma unroll
  for (int i = 0; i < 12; ++i) {
    const int off = tid * 16 + i * 4096;
    unsigned lds_addr = (unsigned)(unsigned long long)((const char*)ldst + off);
    const char* g = (const char*)gsrc + off;
    asm volatile("global_load_async_to_lds_b128 %0, %1, off"
                 :: "v"(lds_addr), "v"(g)
                 : "memory");
  }
}
__device__ __forceinline__ void wait_async0() {
  asm volatile("s_wait_asynccnt 0x0" ::: "memory");
}

// ---------------------------------------------------------------------------
// K1: weights -> bf16 transposed: Wt[col][k] = W[k][col]
// ---------------------------------------------------------------------------
__global__ void cvt_w_kernel(const float* W0, const float* W1,
                             const float* W2, const float* W3,
                             __bf16* T0, __bf16* T1, __bf16* T2, __bf16* T3) {
  const float* W; __bf16* T;
  switch (blockIdx.y) {
    case 0:  W = W0; T = T0; break;
    case 1:  W = W1; T = T1; break;
    case 2:  W = W2; T = T2; break;
    default: W = W3; T = T3; break;
  }
  int idx = blockIdx.x * blockDim.x + threadIdx.x;   // idx = col*D + k
  if (idx < DIMD * DIMD) {
    int col = idx / DIMD, k = idx % DIMD;
    T[idx] = (__bf16)W[(long)k * DIMD + col];
  }
}

// K1b: x (f32 row-major) -> bf16 row-major
__global__ void cvt_x_kernel(const float* __restrict__ x, __bf16* __restrict__ xb) {
  int idx = blockIdx.x * blockDim.x + threadIdx.x;
  if (idx < BN * DIMD) xb[idx] = (__bf16)x[idx];
}

// ---------------------------------------------------------------------------
// K2: Qb = x@Wq+bq, Kb = x@Wk+bk, V stored as Vtb[b][h][dh][n], all bf16.
// One wave per 16x16 output tile. grid = (BN/16, D/16, 3)
// ---------------------------------------------------------------------------
__global__ void qkv_kernel(const __bf16* __restrict__ xb,
                           const __bf16* __restrict__ Wqt,
                           const __bf16* __restrict__ Wkt,
                           const __bf16* __restrict__ Wvt,
                           const float* __restrict__ bq,
                           const float* __restrict__ bk,
                           const float* __restrict__ bv,
                           __bf16* __restrict__ Qb, __bf16* __restrict__ Kb,
                           __bf16* __restrict__ Vtb) {
  const int lane = threadIdx.x;
  const int rb = blockIdx.x * 16;
  const int cb = blockIdx.y * 16;
  const int which = blockIdx.z;
  const __bf16* Wt  = (which == 0) ? Wqt : (which == 1) ? Wkt : Wvt;
  const float* bias = (which == 0) ? bq  : (which == 1) ? bk  : bv;

  v8f acc = {};
#pragma unroll 4
  for (int kc = 0; kc < KCH; ++kc) {
    v16bf a = load_frag(xb, (long)DIMD, rb, kc * 32, lane);
    v16bf b = load_frag(Wt, (long)DIMD, cb, kc * 32, lane);
    acc = wmma_bf16(a, b, acc);
  }
  const int c  = cb + (lane & 15);
  const float bi = bias[c];
  const int m0 = (lane >> 4) * 8;
  if (which == 2) {
    const int h = c >> 6, dh = c & 63;
#pragma unroll
    for (int i = 0; i < 8; ++i) {
      int r = rb + m0 + i, bb = r >> 8, n = r & 255;
      Vtb[(((long)bb * HEADS + h) * DH + dh) * SEQ + n] = (__bf16)(acc[i] + bi);
    }
  } else {
    __bf16* Out = (which == 0) ? Qb : Kb;
#pragma unroll
    for (int i = 0; i < 8; ++i)
      Out[(long)(rb + m0 + i) * DIMD + c] = (__bf16)(acc[i] + bi);
  }
}

// ---------------- attention kernel LDS layout (dynamic, 276KB) -------------
#define SM_STAGE0  0                       // f32  16x768  async buf A  49152
#define SM_STAGE1  49152                   // f32  16x768  async buf B  49152
#define SM_SIMA    98304                   // bf16 16x768  sim tile     24576
#define SM_QT      122880                  // bf16 16x768  Q tile       24576
#define SM_KV      147456                  // f32  16x768  ksim/vsim    49152
#define SM_LSIM    196608                  // f32  16x12x16 sim logits  12288
#define SM_S       208896                  // f32  16x256  self logits  16384
#define SM_P       225280                  // bf16 16x256  softmaxed P   8192
#define SM_O       233472                  // f32  16x768  output acc   49152
#define SMEM_BYTES 282624                  // <= 320KB/WGP

// Project one 16x768 sim tile (bf16 LDS) through Wt (+bias) -> f32 LDS.
// 8 waves, each owns 6 of 48 column tiles; 24 WMMAs per tile.
__device__ __forceinline__ void sim_project(const __bf16* simA, const __bf16* Wt,
                                            const float* bias, float* outLds,
                                            int wave, int lane) {
  for (int ct = wave; ct < DIMD / 16; ct += 8) {
    v8f acc = {};
#pragma unroll 4
    for (int kc = 0; kc < KCH; ++kc) {
      v16bf a = load_frag(simA, (long)DIMD, 0, kc * 32, lane);
      v16bf b = load_frag(Wt,   (long)DIMD, ct * 16, kc * 32, lane);
      acc = wmma_bf16(a, b, acc);
    }
    const int c  = ct * 16 + (lane & 15);
    const float bi = bias[c];
    const int m0 = (lane >> 4) * 8;
#pragma unroll
    for (int i = 0; i < 8; ++i) outLds[(m0 + i) * DIMD + c] = acc[i] + bi;
  }
}

// ---------------------------------------------------------------------------
// K3: fused attention. One 256-thread WG per 16-query tile (grid = 256).
// ---------------------------------------------------------------------------
__global__ void attn_kernel(const float* __restrict__ sim,
                            const __bf16* __restrict__ Qb,
                            const __bf16* __restrict__ Kb,
                            const __bf16* __restrict__ Vtb,
                            const __bf16* __restrict__ Wkt,
                            const float* __restrict__ bk,
                            const __bf16* __restrict__ Wvt,
                            const float* __restrict__ bv,
                            __bf16* __restrict__ AOb) {
  extern __shared__ char smem[];
  float*  stage0 = (float*) (smem + SM_STAGE0);
  float*  stage1 = (float*) (smem + SM_STAGE1);
  __bf16* simA   = (__bf16*)(smem + SM_SIMA);
  __bf16* qtile  = (__bf16*)(smem + SM_QT);
  float*  kv     = (float*) (smem + SM_KV);
  float*  lsim   = (float*) (smem + SM_LSIM);
  float*  S      = (float*) (smem + SM_S);
  __bf16* P      = (__bf16*)(smem + SM_P);
  float*  O      = (float*) (smem + SM_O);

  const int tid  = threadIdx.x;
  const int lane = tid & 31;
  const int wave = tid >> 5;
  const int r0   = blockIdx.x * 16;     // global query-row base
  const int bb   = r0 / SEQ;            // batch (tiles never cross batch)
  const float scale = 0.125f;           // Dh^-0.5
  const float* simbase = sim + (long)r0 * MM * DIMD;

  // ---- phase 0: stage Q tile (bf16 copy) and clear output accumulator ----
  const __bf16* Qrow = Qb + (long)r0 * DIMD;
  for (int i = tid; i < 16 * DIMD; i += 256) {
    qtile[i] = Qrow[i];
    O[i] = 0.f;
  }

  // ---- phase 1: per query, k_sim = sim @ Wk (WMMA) -> sim logits ----------
  async_stage_tile(simbase, stage0, tid);           // q = 0 in flight
  for (int q = 0; q < 16; ++q) {
    float* cur = (q & 1) ? stage1 : stage0;
    float* nxt = (q & 1) ? stage0 : stage1;
    wait_async0();
    __syncthreads();                                // tile q visible WG-wide
    if (q + 1 < 16) async_stage_tile(simbase + (long)(q + 1) * MM * DIMD, nxt, tid);
    for (int i = tid; i < MM * DIMD; i += 256) simA[i] = (__bf16)cur[i];
    __syncthreads();
    sim_project(simA, Wkt, bk, kv, wave, lane);
    __syncthreads();
    if (tid < HEADS * MM) {                // 192 dot products of length 64
      int h = tid >> 4, m = tid & 15;
      const __bf16* qrow = qtile + q * DIMD + h * DH;
      const float*  krow = kv    + m * DIMD + h * DH;
      float s = 0.f;
#pragma unroll 8
      for (int d = 0; d < DH; ++d) s += (float)qrow[d] * krow[d];
      lsim[(q * HEADS + h) * MM + m] = s * scale;
    }
    __syncthreads();
  }

  // ---- phase 2: per head: S = Q K^T (WMMA), softmax(272), O = P V (WMMA) --
  for (int h = 0; h < HEADS; ++h) {
    for (int kt = wave; kt < SEQ / 16; kt += 8) {        // 16 key tiles
      v8f acc = {};
#pragma unroll
      for (int kc = 0; kc < 2; ++kc) {                   // Dh = 2 x 32
        v16bf a = load_frag(qtile, (long)DIMD, 0, h * DH + kc * 32, lane);
        v16bf b = load_frag(Kb + (long)bb * SEQ * DIMD, (long)DIMD,
                            kt * 16, h * DH + kc * 32, lane);
        acc = wmma_bf16(a, b, acc);
      }
      const int key = kt * 16 + (lane & 15);
      const int m0  = (lane >> 4) * 8;
#pragma unroll
      for (int i = 0; i < 8; ++i) S[(m0 + i) * SEQ + key] = acc[i] * scale;
    }
    __syncthreads();

    {  // softmax over 272: 16 groups of 16 lanes, one group per query row
      const int grp = tid >> 4, gl = tid & 15;
      float* srow = S + grp * SEQ;
      float* lrow = lsim + (grp * HEADS + h) * MM;
      float mx = lrow[gl];                        // gl in [0,16) == MM
      for (int k2 = gl; k2 < SEQ; k2 += 16) mx = fmaxf(mx, srow[k2]);
#pragma unroll
      for (int o = 8; o; o >>= 1) mx = fmaxf(mx, __shfl_xor(mx, o, 16));
      float sum = 0.f;
      for (int k2 = gl; k2 < SEQ; k2 += 16) {
        float e = __expf(srow[k2] - mx); srow[k2] = e; sum += e;
      }
      float esim = __expf(lrow[gl] - mx); sum += esim;
#pragma unroll
      for (int o = 8; o; o >>= 1) sum += __shfl_xor(sum, o, 16);
      const float inv = 1.f / sum;
      for (int k2 = gl; k2 < SEQ; k2 += 16) P[grp * SEQ + k2] = (__bf16)(srow[k2] * inv);
      lrow[gl] = esim * inv;                      // keep for phase 3
    }
    __syncthreads();

    if (wave < 4) {                        // O_h (16x64) = P (16x256) @ V_h
      v8f acc = {};
#pragma unroll 2
      for (int kc = 0; kc < SEQ / 32; ++kc) {
        v16bf a = load_frag(P, (long)SEQ, 0, kc * 32, lane);
        v16bf b = load_frag(Vtb + ((long)bb * HEADS + h) * DH * SEQ, (long)SEQ,
                            wave * 16, kc * 32, lane);
        acc = wmma_bf16(a, b, acc);
      }
      const int c  = h * DH + wave * 16 + (lane & 15);
      const int m0 = (lane >> 4) * 8;
#pragma unroll
      for (int i = 0; i < 8; ++i) O[(m0 + i) * DIMD + c] = acc[i];
    }
    __syncthreads();
  }

  // ---- phase 3: per query, v_sim = sim @ Wv (WMMA), O += attn_sim @ v_sim -
  async_stage_tile(simbase, stage0, tid);
  for (int q = 0; q < 16; ++q) {
    float* cur = (q & 1) ? stage1 : stage0;
    float* nxt = (q & 1) ? stage0 : stage1;
    wait_async0();
    __syncthreads();
    if (q + 1 < 16) async_stage_tile(simbase + (long)(q + 1) * MM * DIMD, nxt, tid);
    for (int i = tid; i < MM * DIMD; i += 256) simA[i] = (__bf16)cur[i];
    __syncthreads();
    sim_project(simA, Wvt, bv, kv, wave, lane);
    __syncthreads();
    for (int d = tid; d < DIMD; d += 256) {
      const int h = d >> 6;
      const float* aw = lsim + (q * HEADS + h) * MM;
      float s = 0.f;
#pragma unroll
      for (int m2 = 0; m2 < MM; ++m2) s += aw[m2] * kv[m2 * DIMD + d];
      O[q * DIMD + d] += s;
    }
    __syncthreads();
  }

  // ---- phase 4: write attention output (bf16 for the proj GEMM) ----
  __bf16* AOrow = AOb + (long)r0 * DIMD;
  for (int i = tid; i < 16 * DIMD; i += 256) AOrow[i] = (__bf16)O[i];
}

// ---------------------------------------------------------------------------
// K4: out = AO @ Wp + bp (f32 out).  grid = (BN/16, D/16), 1 wave per tile.
// ---------------------------------------------------------------------------
__global__ void proj_kernel(const __bf16* __restrict__ A,
                            const __bf16* __restrict__ Wt,
                            const float* __restrict__ bias,
                            float* __restrict__ Cm) {
  const int lane = threadIdx.x;
  const int rb = blockIdx.x * 16;
  const int cb = blockIdx.y * 16;
  v8f acc = {};
#pragma unroll 4
  for (int kc = 0; kc < KCH; ++kc) {
    v16bf a = load_frag(A,  (long)DIMD, rb, kc * 32, lane);
    v16bf b = load_frag(Wt, (long)DIMD, cb, kc * 32, lane);
    acc = wmma_bf16(a, b, acc);
  }
  const int c  = cb + (lane & 15);
  const float bi = bias[c];
  const int m0 = (lane >> 4) * 8;
#pragma unroll
  for (int i = 0; i < 8; ++i)
    Cm[(long)(rb + m0 + i) * DIMD + c] = acc[i] + bi;
}

// ---------------------------------------------------------------------------
extern "C" void kernel_launch(void* const* d_in, const int* in_sizes, int n_in,
                              void* d_out, int out_size, void* d_ws, size_t ws_size,
                              hipStream_t stream) {
  const float* x   = (const float*)d_in[0];
  const float* sim = (const float*)d_in[1];
  const float* Wq  = (const float*)d_in[2];
  const float* bq  = (const float*)d_in[3];
  const float* Wk  = (const float*)d_in[4];
  const float* bk  = (const float*)d_in[5];
  const float* Wv  = (const float*)d_in[6];
  const float* bv  = (const float*)d_in[7];
  const float* Wp  = (const float*)d_in[8];
  const float* bp  = (const float*)d_in[9];
  float* out = (float*)d_out;

  char* ws = (char*)d_ws;
  size_t off = 0;
  auto carve = [&](size_t bytes) -> void* {
    off = (off + 255) & ~(size_t)255;
    void* p = ws + off;
    off += bytes;
    return p;
  };
  __bf16* Wqt = (__bf16*)carve((size_t)DIMD * DIMD * 2);
  __bf16* Wkt = (__bf16*)carve((size_t)DIMD * DIMD * 2);
  __bf16* Wvt = (__bf16*)carve((size_t)DIMD * DIMD * 2);
  __bf16* Wpt = (__bf16*)carve((size_t)DIMD * DIMD * 2);
  __bf16* xb  = (__bf16*)carve((size_t)BN * DIMD * 2);
  __bf16* Qb  = (__bf16*)carve((size_t)BN * DIMD * 2);
  __bf16* Kb  = (__bf16*)carve((size_t)BN * DIMD * 2);
  __bf16* Vtb = (__bf16*)carve((size_t)BN * DIMD * 2);   // (b,h,dh,n)
  __bf16* AOb = (__bf16*)carve((size_t)BN * DIMD * 2);
  (void)in_sizes; (void)n_in; (void)out_size; (void)ws_size;

  // K1: weights -> bf16 transposed; x -> bf16
  dim3 cg((DIMD * DIMD + 255) / 256, 4);
  cvt_w_kernel<<<cg, 256, 0, stream>>>(Wq, Wk, Wv, Wp, Wqt, Wkt, Wvt, Wpt);
  cvt_x_kernel<<<(BN * DIMD + 255) / 256, 256, 0, stream>>>(x, xb);

  // K2: QKV projection (WMMA, bf16 outputs)
  dim3 qg(BN / 16, DIMD / 16, 3);
  qkv_kernel<<<qg, 32, 0, stream>>>(xb, Wqt, Wkt, Wvt, bq, bk, bv, Qb, Kb, Vtb);

  // K3: fused sim-projection + attention (WMMA + async LDS double-buffer)
  attn_kernel<<<BN / 16, 256, SMEM_BYTES, stream>>>(sim, Qb, Kb, Vtb,
                                                    Wkt, bk, Wvt, bv, AOb);

  // K4: output projection (WMMA)
  dim3 pg(BN / 16, DIMD / 16);
  proj_kernel<<<pg, 32, 0, stream>>>(AOb, Wpt, bp, out);
}
